// SparseLinear_81243601371170
// MI455X (gfx1250) — compile-verified, matrix operand order checked by reading, not run on hardware
//
#include <hip/hip_runtime.h>

// Problem constants (match reference).
#define N_ROWS 16384
#define IN_F   4096
#define OUT_F  4096

// GEMM tiling: 256x128 workgroup tile, K-step 32, 8 waves of 32 (wave32),
// each wave owns a 64x64 sub-tile = 4x4 WMMA 16x16 tiles
// (16 WMMAs per 8 LDS fragment loads -> high math/LDS-byte density).
#define BM 256
#define BN 128
#define BK 32
#define LDSS 36   // padded LDS row stride in halfs: 32 data + 4 pad -> 72B rows

typedef _Float16 v16h __attribute__((ext_vector_type(16)));
typedef float    v8f  __attribute__((ext_vector_type(8)));
typedef unsigned int       u32;
typedef unsigned long long u64;
typedef u32 v4u_t __attribute__((ext_vector_type(4)));
typedef int v8i_t __attribute__((ext_vector_type(8)));
typedef int v4i_t __attribute__((ext_vector_type(4)));

// ---------------------------------------------------------------------------
// TDM descriptor build + issue (CDNA5 tensor_load_to_lds, ISA ch.8).
// 2D tensor, data_size=2B, tile = BK x rows, LDS padding: after every
// 16 DWORDs (64B = one 32-half row) insert 2 DWORDs (8B) -> 72B row stride,
// matching the LDSS=36 padded LDS layout (bank-conflict-free fragment reads).
// 6-arg builtin form on this toolchain:
//   (uint32x4 g0, int32x8 g1, int32x4 g2, int32x4 g3, int32x8 extra, i32 cpol)
// ---------------------------------------------------------------------------
__device__ __forceinline__ void tdm_load_tile(u32 lds_byte, const _Float16* gsrc,
                                              u32 dim0, u32 dim1, u32 stride0,
                                              u32 rows) {
  u64 ga = (u64)(size_t)gsrc;
  v4u_t g0;
  g0[0] = 1u;                                             // count=1, user desc
  g0[1] = lds_byte;                                       // lds_addr
  g0[2] = (u32)ga;                                        // global_addr[31:0]
  g0[3] = (u32)((ga >> 32) & 0x01FFFFFFu) | 0x80000000u;  // addr[56:32] | type=2
  v8i_t g1;
  g1[0] = (int)((1u << 16)        // data_size = 1 (2 bytes)
              | (1u << 20)        // pad_enable
              | (3u << 22)        // pad_interval code 3 = 16 DWORDs
              | (1u << 25));      // pad_amount  code 1 = 2 DWORDs
  g1[1] = (int)((dim0 & 0xFFFFu) << 16);                              // tensor_dim0 lo
  g1[2] = (int)(((dim0 >> 16) & 0xFFFFu) | ((dim1 & 0xFFFFu) << 16)); // dim0 hi | dim1 lo
  g1[3] = (int)(((dim1 >> 16) & 0xFFFFu) | ((u32)BK << 16));          // dim1 hi | tile_dim0=32
  g1[4] = (int)rows;                                                  // tile_dim1=rows, tile_dim2=0
  g1[5] = (int)stride0;                                               // tensor_dim0_stride lo32
  g1[6] = 0;
  g1[7] = 0;
  v4i_t z4 = {0, 0, 0, 0};
  v8i_t z8 = {0, 0, 0, 0, 0, 0, 0, 0};
  __builtin_amdgcn_tensor_load_to_lds(g0, g1, z4, z4, z8, 0);
}

// ---------------------------------------------------------------------------
// Fragment load from padded LDS tile [rows][LDSS] halfs.
// 16-bit A-operand layout (ISA 7.12.2): lane L holds row M = L%16;
// VGPR j (j<4): K = (L/16)*8 + 2j {+1};  VGPR j (j>=4): K = 16 + same.
// B operand (rows = output columns, K-contiguous) uses the same layout.
// ---------------------------------------------------------------------------
__device__ __forceinline__ v16h load_frag(const _Float16* lds, u32 rowBase, u32 lane) {
  union { v16h h; u32 u[8]; } f;
  const u32 r  = rowBase + (lane & 15u);
  const u32 hf = (lane >> 4) & 1u;
  const _Float16* p = lds + r * LDSS;
#pragma unroll
  for (int j = 0; j < 8; ++j) {
    u32 kk = ((u32)(j >> 2) << 4) | (hf << 3) | ((u32)(j & 3) << 1);
    f.u[j] = *(const u32*)(p + kk);
  }
  return f.h;
}

// ---------------------------------------------------------------------------
// Pass 1a: fp32 spikes -> f16, row-major [N_ROWS][IN_F]. Exact (values are 0/1).
// ---------------------------------------------------------------------------
__global__ __launch_bounds__(256) void cvt_spikes_f16(const float* __restrict__ src,
                                                      _Float16* __restrict__ dst) {
  size_t i = ((size_t)blockIdx.x * blockDim.x + threadIdx.x) * 8;
  float4 a = *(const float4*)(src + i);
  float4 b = *(const float4*)(src + i + 4);
  union { _Float16 h[8]; uint4 u; } p;
  p.h[0] = (_Float16)a.x; p.h[1] = (_Float16)a.y;
  p.h[2] = (_Float16)a.z; p.h[3] = (_Float16)a.w;
  p.h[4] = (_Float16)b.x; p.h[5] = (_Float16)b.y;
  p.h[6] = (_Float16)b.z; p.h[7] = (_Float16)b.w;
  *(uint4*)(dst + i) = p.u;
}

// ---------------------------------------------------------------------------
// Pass 1b: fp32 weight [IN_F][OUT_F] -> f16 transposed [OUT_F][IN_F]
// (LDS-tiled 32x32 transpose: coalesced reads and writes).
// ---------------------------------------------------------------------------
__global__ __launch_bounds__(256) void cvt_weight_T_f16(const float* __restrict__ W,
                                                        _Float16* __restrict__ Bt) {
  __shared__ float t[32][33];
  const u32 k0 = blockIdx.x * 32u;
  const u32 m0 = blockIdx.y * 32u;
  const u32 tx = threadIdx.x;   // 0..31
  const u32 ty = threadIdx.y;   // 0..7
#pragma unroll
  for (u32 j = 0; j < 32; j += 8)
    t[ty + j][tx] = W[(size_t)(k0 + ty + j) * OUT_F + m0 + tx];
  __syncthreads();
#pragma unroll
  for (u32 j = 0; j < 32; j += 8)
    Bt[(size_t)(m0 + ty + j) * IN_F + k0 + tx] = (_Float16)t[tx][ty + j];
}

// ---------------------------------------------------------------------------
// Pass 2: GEMM  out[N][OUT_F] = A_h[N][K] * Bt_h[OUT_F][K]^T + bias
// TDM double-buffered LDS, f16 WMMA, f32 accumulation.
// ---------------------------------------------------------------------------
__global__ __launch_bounds__(256) void snn_gemm_wmma(const _Float16* __restrict__ A,
                                                     const _Float16* __restrict__ Bt,
                                                     const float* __restrict__ bias,
                                                     float* __restrict__ out) {
  __shared__ _Float16 As[2][BM][LDSS];   // 2 * 256*36*2B = 36 KB
  __shared__ _Float16 Bs[2][BN][LDSS];   // 2 * 128*36*2B = 18 KB

  const u32 tid  = threadIdx.x;
  const u32 lane = tid & 31u;
  const u32 wid  = tid >> 5;     // 8 waves
  const u32 wm   = wid & 3u;     // wave row group: 4 x 64 rows
  const u32 wn   = wid >> 2;     // wave col group: 2 x 64 cols

  const u32 rowBlock = blockIdx.y * BM;
  const u32 colBlock = blockIdx.x * BN;

  const _Float16* aTile = A  + (size_t)rowBlock * IN_F;
  const _Float16* bTile = Bt + (size_t)colBlock * IN_F;

  const u32 asAddr[2] = { (u32)(size_t)(const void*)&As[0][0][0],
                          (u32)(size_t)(const void*)&As[1][0][0] };
  const u32 bsAddr[2] = { (u32)(size_t)(const void*)&Bs[0][0][0],
                          (u32)(size_t)(const void*)&Bs[1][0][0] };

  v8f acc[4][4];
#pragma unroll
  for (int mi = 0; mi < 4; ++mi)
#pragma unroll
    for (int ni = 0; ni < 4; ++ni)
      acc[mi][ni] = (v8f)(0.0f);

  const int KT = IN_F / BK;   // 128 K-steps

  // Prologue: fill buffer 0.
  if (wid == 0) {
    tdm_load_tile(asAddr[0], aTile, IN_F, N_ROWS, IN_F, BM);
    tdm_load_tile(bsAddr[0], bTile, IN_F, OUT_F, IN_F, BN);
    __builtin_amdgcn_s_wait_tensorcnt(0);
  }
  __syncthreads();

  for (int kt = 0; kt < KT; ++kt) {
    const int cur = kt & 1;
    const int nxt = cur ^ 1;

    // Prefetch next K-step into the other buffer (TDM, overlaps with math).
    if ((kt + 1 < KT) && wid == 0) {
      const _Float16* aN = aTile + (size_t)(kt + 1) * BK;
      const _Float16* bN = bTile + (size_t)(kt + 1) * BK;
      tdm_load_tile(asAddr[nxt], aN, IN_F, N_ROWS, IN_F, BM);
      tdm_load_tile(bsAddr[nxt], bN, IN_F, OUT_F, IN_F, BN);
    }

    // Compute on the current buffer: 8 fragment loads feed 16 WMMAs per wave.
    v16h af[4], bf[4];
#pragma unroll
    for (int mi = 0; mi < 4; ++mi)
      af[mi] = load_frag(&As[cur][0][0], wm * 64u + (u32)mi * 16u, lane);
#pragma unroll
    for (int ni = 0; ni < 4; ++ni)
      bf[ni] = load_frag(&Bs[cur][0][0], wn * 64u + (u32)ni * 16u, lane);
#pragma unroll
    for (int mi = 0; mi < 4; ++mi)
#pragma unroll
      for (int ni = 0; ni < 4; ++ni)
        acc[mi][ni] = __builtin_amdgcn_wmma_f32_16x16x32_f16(
            false, af[mi], false, bf[ni], (short)0, acc[mi][ni], false, false);

    if ((kt + 1 < KT) && wid == 0)
      __builtin_amdgcn_s_wait_tensorcnt(0);
    __syncthreads();
  }

  // Epilogue: f32 C/D layout -> lane L: N = L&15, M = vgpr + 8*(L>=16).
  const u32 nIn  = lane & 15u;
  const u32 mOff = (lane >> 4) << 3;
#pragma unroll
  for (int mi = 0; mi < 4; ++mi) {
#pragma unroll
    for (int ni = 0; ni < 4; ++ni) {
      const u32 gcol = colBlock + wn * 64u + (u32)ni * 16u + nIn;
      const float bv = bias[gcol];
#pragma unroll
      for (int v = 0; v < 8; ++v) {
        const u32 grow = rowBlock + wm * 64u + (u32)mi * 16u + mOff + (u32)v;
        out[(size_t)grow * OUT_F + gcol] = acc[mi][ni][v] + bv;
      }
    }
  }
}

extern "C" void kernel_launch(void* const* d_in, const int* in_sizes, int n_in,
                              void* d_out, int out_size, void* d_ws, size_t ws_size,
                              hipStream_t stream) {
  const float* spikes = (const float*)d_in[0];   // [N_ROWS, IN_F] fp32 (0/1)
  const float* weight = (const float*)d_in[1];   // [IN_F, OUT_F] fp32
  const float* bias   = (const float*)d_in[2];   // [OUT_F] fp32
  float* out          = (float*)d_out;           // [N_ROWS, OUT_F] fp32

  _Float16* wsA = (_Float16*)d_ws;                       // 128 MB: spikes f16
  _Float16* wsB = wsA + (size_t)N_ROWS * IN_F;           //  32 MB: weight^T f16

  // Pass 1: precision conversion (one-shot, ~400MB of traffic ≈ 17us at 23.3TB/s).
  {
    const size_t n = (size_t)N_ROWS * IN_F;              // 67.1M, multiple of 8*256
    cvt_spikes_f16<<<dim3((unsigned)(n / (8 * 256))), dim3(256), 0, stream>>>(spikes, wsA);
    dim3 tg(IN_F / 32, OUT_F / 32), tb(32, 8);
    cvt_weight_T_f16<<<tg, tb, 0, stream>>>(weight, wsB);
  }

  // Pass 2: compute-bound WMMA GEMM.
  dim3 grid(OUT_F / BN, N_ROWS / BM);                    // (32, 64)
  snn_gemm_wmma<<<grid, dim3(256), 0, stream>>>(wsA, wsB, bias, out);
}
